// VNMaxPool_31035433681604
// MI455X (gfx1250) — matrix operand order repaired and checked by reading, not compiled, not run on hardware
//
#include <hip/hip_runtime.h>

// ---------------------------------------------------------------------------
// VNMaxPool for MI455X (gfx1250, wave32, WMMA)
//   d[b,c,k,n] = sum_i W[c,i] x[b,i,k,n]   (bf16x3 split WMMA, ~fp32 accuracy)
//   dot[b,c,n] = sum_k x[b,c,k,n]*d[b,c,k,n]
//   idx = argmax_n dot ; out[b,c,:] = x[b,c,:,idx]
// ---------------------------------------------------------------------------

typedef __attribute__((ext_vector_type(16))) __bf16 v16bf;
typedef __attribute__((ext_vector_type(8)))  float  v8f;

#define B_       16
#define C_       256
#define N_       4096
#define NCOLS    32               // columns of X staged in LDS per chunk
#define CSTRIDE  272              // 256 K-values + 16 pad (544B: 32B-aligned, breaks bank stride)
#define STRIPS   32               // workgroups along n per batch
#define COLS_WG  (N_ / STRIPS)    // 128 columns per workgroup
#define CHUNKS   (COLS_WG / NCOLS)

// Order-preserving map f32 -> u32 (monotone under unsigned compare)
__device__ __forceinline__ unsigned monoKey(float f) {
    unsigned u = __float_as_uint(f);
    return (u & 0x80000000u) ? ~u : (u | 0x80000000u);
}

// ---------------------------------------------------------------------------
// Kernel 1: split W (f32, [C,C]) into hi/lo bf16 stored in WMMA A-fragment
// order: frag[cb][t][lane][j], j = 16 contiguous bf16 per lane (32B loads).
// A-matrix 16x32 layout (contiguous K, matches B): lane<16 : M=lane,    K=j
//                                                  lane>=16: M=lane-16, K=j+16
// ---------------------------------------------------------------------------
__global__ void wsplit_kernel(const float* __restrict__ W,
                              __bf16* __restrict__ Whi, __bf16* __restrict__ Wlo) {
    int tid = blockIdx.x * 256 + threadIdx.x;          // 65536 total
    int j  = tid & 15;
    int l  = (tid >> 4) & 31;
    int t  = (tid >> 9) & 7;
    int cb = tid >> 12;
    int c  = cb * 16 + (l & 15);
    int K  = 32 * t + j + ((l < 16) ? 0 : 16);
    float w  = W[c * C_ + K];
    __bf16 hi = (__bf16)w;
    __bf16 lo = (__bf16)(w - (float)hi);
    Whi[tid] = hi;
    Wlo[tid] = lo;
}

// ---------------------------------------------------------------------------
// Kernel 2: fused GEMM (bf16x3 WMMA) + dot + per-(b,c) argmax via atomicMax.
// Block = 512 threads = 16 waves; wave w owns c-block w (rows 16w..16w+15).
// Grid = (STRIPS, B). All 16 c-blocks share one LDS-staged X strip -> x read
// from HBM exactly once.
// ---------------------------------------------------------------------------
__global__ __launch_bounds__(512)
void fused_kernel(const float* __restrict__ x,
                  const __bf16* __restrict__ Whi, const __bf16* __restrict__ Wlo,
                  unsigned long long* __restrict__ best) {
    __shared__ __align__(32) __bf16 Xh[3 * NCOLS * CSTRIDE];
    __shared__ __align__(32) __bf16 Xl[3 * NCOLS * CSTRIDE];

    const int b    = blockIdx.y;
    const int tid  = threadIdx.x;
    const int wave = tid >> 5;          // c-block index 0..15
    const int lane = tid & 31;
    const int half = lane >> 4;         // 0: rows r..r ; 1: rows r+8 (C/D layout)
    const int nl   = lane & 15;         // column-in-tile for B/C/D operands

    // A fragments, register resident, reused across the whole 128-col strip.
    v16bf Ahi[8], Alo[8];
#pragma unroll
    for (int t = 0; t < 8; ++t) {
        int off = ((wave * 8 + t) * 32 + lane) * 16;    // 32B per lane, coalesced
        Ahi[t] = *(const v16bf*)(Whi + off);
        Alo[t] = *(const v16bf*)(Wlo + off);
    }

    float bval[8];
    int   bidx[8];
#pragma unroll
    for (int r = 0; r < 8; ++r) { bval[r] = -__builtin_inff(); bidx[r] = 0; }

    const int n_base = blockIdx.x * COLS_WG;

    for (int chunk = 0; chunk < CHUNKS; ++chunk) {
        const int n0 = n_base + chunk * NCOLS;

        // Stage X chunk (3k x 256i x 32n) -> LDS as bf16 hi/lo, K-contiguous
        // per column. Global reads: 128B contiguous per 32-thread group.
        for (int idx = tid; idx < 3 * C_ * NCOLS; idx += 512) {
            int n = idx & (NCOLS - 1);
            int i = (idx >> 5) & 255;
            int k = idx >> 13;                 // NCOLS*C_ = 8192
            float v = x[((size_t)((b * C_ + i) * 3 + k)) * N_ + (n0 + n)];
            __bf16 hi = (__bf16)v;
            __bf16 lo = (__bf16)(v - (float)hi);
            int a = (k * NCOLS + n) * CSTRIDE + i;
            Xh[a] = hi;
            Xl[a] = lo;
        }
        __syncthreads();

        for (int ntile = 0; ntile < NCOLS / 16; ++ntile) {
            const int ncol  = ntile * 16 + nl;       // column within chunk
            const int nglob = n0 + ntile * 16 + nl;  // global n for this lane
            v8f dot = {0.f, 0.f, 0.f, 0.f, 0.f, 0.f, 0.f, 0.f};

            for (int k = 0; k < 3; ++k) {
                v8f acc = {0.f, 0.f, 0.f, 0.f, 0.f, 0.f, 0.f, 0.f};
#pragma unroll
                for (int t = 0; t < 8; ++t) {
                    // B 32x16 layout: lane%16 = N, 16 contiguous K per lane,
                    // lanes>=16 hold the upper K half. 32B-aligned ds loads.
                    const int ba = (k * NCOLS + ncol) * CSTRIDE + 32 * t + 16 * half;
                    v16bf Bh = *(const v16bf*)&Xh[ba];
                    v16bf Bl = *(const v16bf*)&Xl[ba];
                    acc = __builtin_amdgcn_wmma_f32_16x16x32_bf16(
                        false, Ahi[t], false, Bh, (short)0, acc, false, false);
                    acc = __builtin_amdgcn_wmma_f32_16x16x32_bf16(
                        false, Ahi[t], false, Bl, (short)0, acc, false, false);
                    acc = __builtin_amdgcn_wmma_f32_16x16x32_bf16(
                        false, Alo[t], false, Bh, (short)0, acc, false, false);
                }
                // dot += x (exact f32, L2-hot: just streamed) * d
#pragma unroll
                for (int r = 0; r < 8; ++r) {
                    int c = wave * 16 + r + half * 8;
                    float xv = x[((size_t)((b * C_ + c) * 3 + k)) * N_ + nglob];
                    dot[r] += xv * acc[r];
                }
            }
            // per-lane running argmax (first-index tie-break)
#pragma unroll
            for (int r = 0; r < 8; ++r) {
                float v = dot[r];
                if (v > bval[r] || (v == bval[r] && nglob < bidx[r])) {
                    bval[r] = v; bidx[r] = nglob;
                }
            }
        }
        __syncthreads();
    }

    // Reduce across the 16 columns held in each half-wave, then one atomic
    // per (b,c): packed key = (monotone f32 key << 32) | ~n  (lower n wins ties).
#pragma unroll
    for (int r = 0; r < 8; ++r) {
        float v = bval[r];
        int   n = bidx[r];
        for (int m = 1; m < 16; m <<= 1) {
            float ov = __shfl_xor(v, m, 32);
            int   on = __shfl_xor(n, m, 32);
            if (ov > v || (ov == v && on < n)) { v = ov; n = on; }
        }
        if (nl == 0) {
            int c = wave * 16 + r + half * 8;
            unsigned long long packed =
                ((unsigned long long)monoKey(v) << 32) |
                (unsigned long long)(0xFFFFFFFFu - (unsigned)n);
            atomicMax(&best[b * C_ + c], packed);
        }
    }
}

// ---------------------------------------------------------------------------
// Kernel 3: decode argmax index, gather the 3-vector.
// ---------------------------------------------------------------------------
__global__ void gather_kernel(const float* __restrict__ x,
                              const unsigned long long* __restrict__ best,
                              float* __restrict__ out) {
    int id = blockIdx.x * 256 + threadIdx.x;    // id = b*C_ + c
    if (id >= B_ * C_) return;
    unsigned long long p = best[id];
    unsigned n = 0xFFFFFFFFu - (unsigned)(p & 0xFFFFFFFFull);
#pragma unroll
    for (int kk = 0; kk < 3; ++kk)
        out[id * 3 + kk] = x[((size_t)(id * 3 + kk)) * N_ + n];
}

// ---------------------------------------------------------------------------
extern "C" void kernel_launch(void* const* d_in, const int* in_sizes, int n_in,
                              void* d_out, int out_size, void* d_ws, size_t ws_size,
                              hipStream_t stream) {
    (void)in_sizes; (void)n_in; (void)out_size; (void)ws_size;
    const float* x = (const float*)d_in[0];     // [B, C, 3, N] f32
    const float* W = (const float*)d_in[1];     // [C, C] f32
    float* out = (float*)d_out;                 // [B, C, 3] f32

    char* ws = (char*)d_ws;
    __bf16* Whi = (__bf16*)ws;                               // 128 KB
    __bf16* Wlo = (__bf16*)(ws + (size_t)C_ * C_ * 2);       // 128 KB
    unsigned long long* best =
        (unsigned long long*)(ws + (size_t)C_ * C_ * 4);     // 32 KB

    hipMemsetAsync(best, 0, (size_t)B_ * C_ * sizeof(unsigned long long), stream);

    wsplit_kernel<<<C_ * C_ / 256, 256, 0, stream>>>(W, Whi, Wlo);

    dim3 grid(STRIPS, B_);
    fused_kernel<<<grid, 512, 0, stream>>>(x, Whi, Wlo, best);

    gather_kernel<<<(B_ * C_ + 255) / 256, 256, 0, stream>>>(x, best, out);
}
